// LMHSA_38431367365189
// MI455X (gfx1250) — compile-verified, matrix operand order checked by reading, not run on hardware
//
#include <hip/hip_runtime.h>
#include <hip/hip_bf16.h>
#include <math.h>

// ---------------------------------------------------------------------------
// CDNA5 (gfx1250) wave32 WMMA types/helpers
// ---------------------------------------------------------------------------
typedef __bf16 v16bf __attribute__((ext_vector_type(16)));
typedef float  v8f   __attribute__((ext_vector_type(8)));

union FragBF { v16bf v; uint4 q[2]; };

__device__ __forceinline__ unsigned short f2bf(float f) {
    unsigned int u = __float_as_uint(f);
    u += 0x7FFFu + ((u >> 16) & 1u);          // round-to-nearest-even
    return (unsigned short)(u >> 16);
}

__device__ __forceinline__ v8f wmma_bf16(v16bf a, v16bf b, v8f c) {
    // v_wmma_f32_16x16x32_bf16  (neg_a, A, neg_b, B, c_mod, C, reuse_a, reuse_b)
    return __builtin_amdgcn_wmma_f32_16x16x32_bf16(false, a, false, b, (short)0, c, false, false);
}

// 16-bit A/B fragment: lane L holds row/col (L&15); lanes 0-15 K=[k..k+7],[k+16..k+23],
// lanes 16-31 shifted by +8.  Two 16-byte loads per lane.
__device__ __forceinline__ v16bf load_frag(const unsigned short* rowptr, int kbase, int lane) {
    int off = kbase + ((lane & 16) ? 8 : 0);
    FragBF u;
    u.q[0] = *(const uint4*)(rowptr + off);
    u.q[1] = *(const uint4*)(rowptr + off + 16);
    return u.v;
}

// ---------------------------------------------------------------------------
// Problem constants
// ---------------------------------------------------------------------------
#define NB   8
#define NC   256
#define HW   1024
#define IMW  32
#define PADW 34
#define PADSLOTS (PADW * PADW)   // 1156
#define NHEADS 8
#define NTOT (NB * NC * HW)      // 2097152

// ---------------------------------------------------------------------------
// 1) whole-tensor layernorm statistics
// ---------------------------------------------------------------------------
__global__ void k_zero(float* red) { red[0] = 0.f; red[1] = 0.f; }

__global__ void k_clear(unsigned int* __restrict__ p, int n) {
    int i = blockIdx.x * blockDim.x + threadIdx.x;
    if (i < n) p[i] = 0u;
}

__global__ void k_reduce(const float* __restrict__ x, float* __restrict__ red, int n) {
    __shared__ float s1[256], s2[256];
    int tid = threadIdx.x;
    float a = 0.f, b = 0.f;
    for (int i = blockIdx.x * blockDim.x + tid; i < n; i += gridDim.x * blockDim.x) {
        float v = x[i]; a += v; b += v * v;
    }
    s1[tid] = a; s2[tid] = b;
    __syncthreads();
    for (int s = 128; s > 0; s >>= 1) {
        if (tid < s) { s1[tid] += s1[tid + s]; s2[tid] += s2[tid + s]; }
        __syncthreads();
    }
    if (tid == 0) { atomicAdd(&red[0], s1[0]); atomicAdd(&red[1], s2[0]); }
}

// ---------------------------------------------------------------------------
// 2) transpose x[b,c,p] -> pixel-major bf16:
//    xrp : raw, zero-padded 34x34 image (conv input, no boundary masking)
//    xn  : normalized, flat [b,hw,c] (q-projection input)
// ---------------------------------------------------------------------------
__global__ void k_pack(const float* __restrict__ x, const float* __restrict__ red,
                       unsigned short* __restrict__ xn, unsigned short* __restrict__ xrp) {
    __shared__ float tile[32][33];
    int b  = blockIdx.x >> 5;
    int pt = blockIdx.x & 31;
    int c0 = blockIdx.y * 32;
    int p0 = pt * 32;
    int tx = threadIdx.x, ty = threadIdx.y;
#pragma unroll
    for (int i = 0; i < 4; ++i) {
        int c = c0 + ty + i * 8;
        tile[ty + i * 8][tx] = x[((size_t)b * NC + c) * HW + p0 + tx];
    }
    __syncthreads();
    float mean = red[0] * (1.0f / (float)NTOT);
    float var  = red[1] * (1.0f / (float)NTOT) - mean * mean;
    float rstd = rsqrtf(var + 1e-5f);
#pragma unroll
    for (int i = 0; i < 4; ++i) {
        int p = p0 + ty + i * 8;
        float v = tile[tx][ty + i * 8];
        size_t o = ((size_t)b * HW + p) * NC + c0 + tx;
        xn[o] = f2bf((v - mean) * rstd);
        int ps = ((p >> 5) + 1) * PADW + (p & 31) + 1;          // interior of padded image
        xrp[((size_t)b * PADSLOTS + ps) * NC + c0 + tx] = f2bf(v);
    }
}

// ---------------------------------------------------------------------------
// 3) weight repack to bf16 in B-fragment order
// ---------------------------------------------------------------------------
__global__ void k_wtrans(const float* __restrict__ W, unsigned short* __restrict__ out) {
    int o = blockIdx.x, i = threadIdx.x;                 // out[out_ch][in_ch]
    out[(size_t)o * NC + i] = f2bf(W[(size_t)i * NC + o]);
}

__global__ void k_wconv(const float* __restrict__ W, unsigned short* __restrict__ out) {
    int t = blockIdx.x >> 8, co = blockIdx.x & 255, ci = threadIdx.x;   // out[t][co][ci]
    out[((size_t)t * NC + co) * NC + ci] = f2bf(W[(size_t)co * (NC * 9) + ci * 9 + t]);
}

// ---------------------------------------------------------------------------
// 4) conv3x3 as 9-tap accumulated GEMM (bf16 WMMA, f32 acc), bias epilogue
//    block = 256 thr (8 waves); wave tile = 32 pixels x 64 out-channels:
//    2 A-fragments share each B-fragment -> 8 WMMA per 12 b128 loads
// ---------------------------------------------------------------------------
__global__ __launch_bounds__(256) void k_conv(const unsigned short* __restrict__ xrp,
                                              const unsigned short* __restrict__ Wc,
                                              const float* __restrict__ bias,
                                              unsigned short* __restrict__ out) {
    int bid  = blockIdx.x;
    int b    = bid >> 4, pt = bid & 15;
    int wave = threadIdx.x >> 5;
    int lane = threadIdx.x & 31;
    int mT   = wave >> 2, nT = wave & 3;
    int p0   = pt * 64 + mT * 32;
    int n0   = nT * 64;
    int mrow = lane & 15, mhi = lane >> 4;

    int pA  = p0 + mrow;
    int pB  = p0 + 16 + mrow;
    int psA = ((pA >> 5) + 1) * PADW + (pA & 31) + 1;
    int psB = ((pB >> 5) + 1) * PADW + (pB & 31) + 1;
    const unsigned short* xb = xrp + (size_t)b * PADSLOTS * NC;

    v8f acc[8] = {};
    for (int t = 0; t < 9; ++t) {
        int toff = (t / 3 - 1) * PADW + (t % 3 - 1);
        const unsigned short* a0p = xb + (size_t)(psA + toff) * NC;
        const unsigned short* a1p = xb + (size_t)(psB + toff) * NC;
        const unsigned short* b0  = Wc + (size_t)t * (NC * NC) + (size_t)(n0 + mrow) * NC;
        for (int k = 0; k < NC; k += 32) {
            __builtin_prefetch(a0p + k + 64, 0, 1);
            v16bf a0 = load_frag(a0p, k, lane);
            v16bf a1 = load_frag(a1p, k, lane);
#pragma unroll
            for (int j = 0; j < 4; ++j) {
                v16bf bf = load_frag(b0 + j * 16 * NC, k, lane);
                acc[j]     = wmma_bf16(a0, bf, acc[j]);
                acc[4 + j] = wmma_bf16(a1, bf, acc[4 + j]);
            }
        }
    }
#pragma unroll
    for (int mi = 0; mi < 2; ++mi) {
#pragma unroll
        for (int j = 0; j < 4; ++j) {
            int col = n0 + j * 16 + mrow;
            float bv = bias[col];
#pragma unroll
            for (int r = 0; r < 8; ++r) {
                int m = mi * 16 + r + mhi * 8;
                out[((size_t)b * HW + p0 + m) * NC + col] = f2bf(acc[mi * 4 + j][r] + bv);
            }
        }
    }
}

// ---------------------------------------------------------------------------
// 5) generic 8192x256x256 projection GEMM, 32x64 wave tiles
//    mode 0: bf16 head-major [b,H,hw,32]   (q, k)
//    mode 2: bf16 transposed [b,H,32,hw]   (v, for attention B-fragments)
//    mode 1: f32 out + bias + flat residual (final o-proj, matches raw reshape)
// ---------------------------------------------------------------------------
__global__ __launch_bounds__(256) void k_proj(const unsigned short* __restrict__ A,
                                              const unsigned short* __restrict__ Wt,
                                              const float* __restrict__ bias,
                                              unsigned short* __restrict__ outbf,
                                              float* __restrict__ outf,
                                              const float* __restrict__ resid,
                                              int mode) {
    int bid  = blockIdx.x;
    int wave = threadIdx.x >> 5, lane = threadIdx.x & 31;
    int mT   = wave >> 2, nT = wave & 3;
    int r0   = bid * 64 + mT * 32;
    int n0   = nT * 64;
    int mrow = lane & 15, mhi = lane >> 4;

    const unsigned short* a0p = A + (size_t)(r0 + mrow) * NC;
    const unsigned short* a1p = A + (size_t)(r0 + 16 + mrow) * NC;
    const unsigned short* b0  = Wt + (size_t)(n0 + mrow) * NC;

    v8f acc[8] = {};
    for (int k = 0; k < NC; k += 32) {
        v16bf a0 = load_frag(a0p, k, lane);
        v16bf a1 = load_frag(a1p, k, lane);
#pragma unroll
        for (int j = 0; j < 4; ++j) {
            v16bf bf = load_frag(b0 + j * 16 * NC, k, lane);
            acc[j]     = wmma_bf16(a0, bf, acc[j]);
            acc[4 + j] = wmma_bf16(a1, bf, acc[4 + j]);
        }
    }
#pragma unroll
    for (int mi = 0; mi < 2; ++mi) {
#pragma unroll
        for (int j = 0; j < 4; ++j) {
            int col = n0 + j * 16 + mrow;
            float bv = bias[col];
#pragma unroll
            for (int r = 0; r < 8; ++r) {
                int row = r0 + mi * 16 + r + mhi * 8;
                float val = acc[mi * 4 + j][r] + bv;
                if (mode == 0) {
                    int bb = row >> 10, pp = row & 1023, h = col >> 5, d = col & 31;
                    outbf[(((size_t)bb * NHEADS + h) * HW + pp) * 32 + d] = f2bf(val);
                } else if (mode == 2) {
                    int bb = row >> 10, pp = row & 1023, h = col >> 5, d = col & 31;
                    outbf[(((size_t)bb * NHEADS + h) * 32 + d) * HW + pp] = f2bf(val);
                } else {
                    size_t idx = (size_t)row * NC + col;
                    outf[idx] = val + resid[idx];
                }
            }
        }
    }
}

// ---------------------------------------------------------------------------
// 6) flash attention: per wave one 16-row q tile over 1024 keys (32/iter),
//    online softmax with additive bias B; never materializes the score matrix
// ---------------------------------------------------------------------------
__global__ __launch_bounds__(128) void k_attn(const unsigned short* __restrict__ qh,
                                              const unsigned short* __restrict__ kh,
                                              const unsigned short* __restrict__ vt,
                                              const float* __restrict__ Bb,
                                              unsigned short* __restrict__ att) {
    __shared__ unsigned short pls[4][16 * 32];   // per-wave P relayout tile
    int bh   = blockIdx.x;
    int b    = bh >> 3, h = bh & 7;
    int wave = threadIdx.x >> 5, lane = threadIdx.x & 31;
    int qt   = blockIdx.y * 4 + wave;
    int q0   = qt * 16;
    int mrow = lane & 15, mhi = lane >> 4;
    unsigned short* pw = pls[wave];

    v16bf aq = load_frag(qh + ((size_t)bh * HW + q0 + mrow) * 32, 0, lane);

    v8f o0 = {}, o1 = {};
    float mrun[8], lrun[8];
#pragma unroll
    for (int r = 0; r < 8; ++r) { mrun[r] = -1e30f; lrun[r] = 0.f; }
    const float scale = 0.17677669529663687f;    // 1/sqrt(32)

    for (int kc = 0; kc < HW; kc += 32) {
        const unsigned short* kb = kh + ((size_t)bh * HW + kc) * 32;
        v16bf bk0 = load_frag(kb + (size_t)mrow * 32,        0, lane);
        v16bf bk1 = load_frag(kb + (size_t)(16 + mrow) * 32, 0, lane);
        v8f zc = {};
        v8f s0 = wmma_bf16(aq, bk0, zc);
        v8f s1 = wmma_bf16(aq, bk1, zc);

        const float* bp = Bb + ((size_t)h * HW + q0) * HW + kc + mrow;
#pragma unroll
        for (int r = 0; r < 8; ++r) {
            int m = r + mhi * 8;
            float v0 = s0[r] * scale + bp[(size_t)m * HW];
            float v1 = s1[r] * scale + bp[(size_t)m * HW + 16];
            float mx = fmaxf(v0, v1);
            for (int s = 1; s < 16; s <<= 1) mx = fmaxf(mx, __shfl_xor(mx, s, 32));
            float nmax = fmaxf(mrun[r], mx);
            float e0 = __expf(v0 - nmax);
            float e1 = __expf(v1 - nmax);
            float rs = e0 + e1;
            for (int s = 1; s < 16; s <<= 1) rs += __shfl_xor(rs, s, 32);
            float alpha = __expf(mrun[r] - nmax);
            lrun[r] = lrun[r] * alpha + rs;
            mrun[r] = nmax;
            o0[r] = o0[r] * alpha;
            o1[r] = o1[r] * alpha;
            pw[m * 32 + mrow]      = f2bf(e0);   // C-layout -> LDS [row][key]
            pw[m * 32 + 16 + mrow] = f2bf(e1);
        }
        // reload P in A-fragment layout (same wave: DS ops stay ordered)
        v16bf ap = load_frag(pw + mrow * 32, 0, lane);
        const unsigned short* vb = vt + ((size_t)bh * 32) * HW + kc;
        v16bf bv0 = load_frag(vb + (size_t)mrow * HW,        0, lane);
        v16bf bv1 = load_frag(vb + (size_t)(16 + mrow) * HW, 0, lane);
        o0 = wmma_bf16(ap, bv0, o0);
        o1 = wmma_bf16(ap, bv1, o1);
    }
#pragma unroll
    for (int r = 0; r < 8; ++r) {
        int m = r + mhi * 8;
        float inv = 1.0f / lrun[r];
        size_t base = ((size_t)b * HW + q0 + m) * NC + h * 32;
        att[base + mrow]      = f2bf(o0[r] * inv);
        att[base + 16 + mrow] = f2bf(o1[r] * inv);
    }
}

// ---------------------------------------------------------------------------
// host launcher
// ---------------------------------------------------------------------------
extern "C" void kernel_launch(void* const* d_in, const int* in_sizes, int n_in,
                              void* d_out, int out_size, void* d_ws, size_t ws_size,
                              hipStream_t stream) {
    const float* x   = (const float*)d_in[0];
    const float* Wq  = (const float*)d_in[1];
    const float* bq  = (const float*)d_in[2];
    const float* ckw = (const float*)d_in[3];
    const float* ckb = (const float*)d_in[4];
    const float* Wk  = (const float*)d_in[5];
    const float* bk  = (const float*)d_in[6];
    const float* cvw = (const float*)d_in[7];
    const float* cvb = (const float*)d_in[8];
    const float* Wv  = (const float*)d_in[9];
    const float* bv  = (const float*)d_in[10];
    const float* Wo  = (const float*)d_in[11];
    const float* bo  = (const float*)d_in[12];
    const float* Bb  = (const float*)d_in[13];
    float* out = (float*)d_out;

    char* w = (char*)d_ws;
    const size_t MB4   = (size_t)NB * HW * NC * sizeof(unsigned short);        // 4 MiB
    const size_t PADSZ = (size_t)NB * PADSLOTS * NC * sizeof(unsigned short);  // ~4.5 MiB
    size_t off = 0;
    float* red = (float*)(w + off);                   off += 256;
    unsigned short* xn  = (unsigned short*)(w + off); off += MB4;
    unsigned short* xrp = (unsigned short*)(w + off); off += PADSZ;
    unsigned short* kc  = (unsigned short*)(w + off); off += MB4;
    unsigned short* vc  = (unsigned short*)(w + off); off += MB4;
    unsigned short* qhd = (unsigned short*)(w + off); off += MB4;
    unsigned short* khd = (unsigned short*)(w + off); off += MB4;
    unsigned short* vtd = (unsigned short*)(w + off); off += MB4;
    unsigned short* att = (unsigned short*)(w + off); off += MB4;
    unsigned short* wqt = (unsigned short*)(w + off); off += (size_t)NC * NC * 2;
    unsigned short* wkt = (unsigned short*)(w + off); off += (size_t)NC * NC * 2;
    unsigned short* wvt = (unsigned short*)(w + off); off += (size_t)NC * NC * 2;
    unsigned short* wot = (unsigned short*)(w + off); off += (size_t)NC * NC * 2;
    unsigned short* wck = (unsigned short*)(w + off); off += (size_t)9 * NC * NC * 2;
    unsigned short* wcv = (unsigned short*)(w + off); off += (size_t)9 * NC * NC * 2;

    k_zero <<<1, 1, 0, stream>>>(red);
    {   // zero the padded conv input (halo stays zero, interior overwritten by k_pack)
        int n = (int)(PADSZ / sizeof(unsigned int));
        k_clear<<<(n + 255) / 256, 256, 0, stream>>>((unsigned int*)xrp, n);
    }
    k_reduce<<<512, 256, 0, stream>>>(x, red, NTOT);
    k_pack  <<<dim3(NB * 32, NC / 32), dim3(32, 8), 0, stream>>>(x, red, xn, xrp);

    k_wtrans<<<NC, NC, 0, stream>>>(Wq, wqt);
    k_wtrans<<<NC, NC, 0, stream>>>(Wk, wkt);
    k_wtrans<<<NC, NC, 0, stream>>>(Wv, wvt);
    k_wtrans<<<NC, NC, 0, stream>>>(Wo, wot);
    k_wconv <<<9 * NC, NC, 0, stream>>>(ckw, wck);
    k_wconv <<<9 * NC, NC, 0, stream>>>(cvw, wcv);

    k_conv<<<NB * 16, 256, 0, stream>>>(xrp, wck, ckb, kc);
    k_conv<<<NB * 16, 256, 0, stream>>>(xrp, wcv, cvb, vc);

    k_proj<<<128, 256, 0, stream>>>(xn, wqt, bq, qhd, nullptr, nullptr, 0);
    k_proj<<<128, 256, 0, stream>>>(kc, wkt, bk, khd, nullptr, nullptr, 0);
    k_proj<<<128, 256, 0, stream>>>(vc, wvt, bv, vtd, nullptr, nullptr, 2);

    k_attn<<<dim3(NB * NHEADS, 16), 128, 0, stream>>>(qhd, khd, vtd, Bb, att);

    k_proj<<<128, 256, 0, stream>>>(att, wot, bo, nullptr, out, x, 1);
}
